// GraphX_11347303596563
// MI455X (gfx1250) — matrix operand order repaired and checked by reading, not compiled, not run on hardware
//
#include <hip/hip_runtime.h>
#include <stdint.h>

// ---------------- problem constants ----------------
#define B_   256
#define FP_  80
#define SP_  160
#define H_   128
#define NH_  4
#define DH_  32
#define NPG_ (FP_ + SP_ + 2)       // 242
#define N_   (B_ * NPG_)           // 61952
#define E_   1000000

typedef __attribute__((ext_vector_type(16))) __bf16 v16bf;
typedef __attribute__((ext_vector_type(8)))  __bf16 v8bf;
typedef __attribute__((ext_vector_type(8)))  float  v8f;

// float -> bf16 bits, round-to-nearest-even
__device__ __forceinline__ unsigned short f2bf(float f) {
    unsigned u = __float_as_uint(f);
    unsigned r = u + 0x7FFFu + ((u >> 16) & 1u);
    return (unsigned short)(r >> 16);
}

// monotone float<->uint encoding for atomic max over signed floats
__device__ __forceinline__ unsigned fenc(float f) {
    unsigned u = __float_as_uint(f);
    return (u & 0x80000000u) ? ~u : (u | 0x80000000u);
}
__device__ __forceinline__ float fdec(unsigned k) {
    return (k & 0x80000000u) ? __uint_as_float(k ^ 0x80000000u)
                             : __uint_as_float(~k);
}

// ---------------- 1) fold W_type @ W_gat -> bf16, stored N-major (transposed) ----
__global__ void combine_weights(const float* __restrict__ Wt,
                                const float* __restrict__ Wg,
                                unsigned short* __restrict__ WcT) {
    int idx = blockIdx.x * blockDim.x + threadIdx.x;   // H*H threads
    int n = idx >> 7, i = idx & (H_ - 1);
    float acc = 0.f;
    #pragma unroll 8
    for (int j = 0; j < H_; ++j) acc += Wt[i * H_ + j] * Wg[j * H_ + n];
    WcT[n * H_ + i] = f2bf(acc);                       // WcT[n][k] = Wc[k][n]
}

// ---------------- 2) z = gather(tab, ids) @ Wc via WMMA bf16 --------------------
__global__ __launch_bounds__(128)
void node_project(const float* __restrict__ tab, const int* __restrict__ ids,
                  const unsigned short* __restrict__ WcT,
                  float* __restrict__ z, int perGraph, int nodeOff) {
    __shared__ unsigned short At[16 * 136];            // 16 rows x 128 bf16, padded
    __shared__ int nodeRow[16];
    const int r0  = blockIdx.x * 16;
    const int tid = threadIdx.x;

    if (tid < 16) {
        int r = r0 + tid;
        nodeRow[tid] = (r / perGraph) * NPG_ + nodeOff + (r % perGraph);
    }
    {   // cooperative gather: thread -> (row = tid/8, 16 cols at (tid%8)*16)
        int lr = tid >> 3;
        int c0 = (tid & 7) * 16;
        long id = (long)ids[r0 + lr];
        const float* src = tab + id * H_ + c0;
        unsigned* dstw = (unsigned*)&At[lr * 136 + c0];
        #pragma unroll
        for (int c = 0; c < 16; c += 2) {
            unsigned lo = f2bf(src[c]);
            unsigned hi = f2bf(src[c + 1]);
            dstw[c >> 1] = lo | (hi << 16);
        }
    }
    __syncthreads();

    const int lane = tid & 31;
    const int half = lane >> 4;     // lane 0-15 vs 16-31
    const int l15  = lane & 15;
    const int n0   = (tid >> 5) * 32;   // each wave owns 32 output cols (2 tiles)

    v8f acc0 = {}; v8f acc1 = {};
    #pragma unroll
    for (int ks = 0; ks < 4; ++ks) {
        // A fragment (16x32 bf16): lanes 0-15 K{0..15}, lanes 16-31 K{16..31}
        const unsigned short* arow = &At[l15 * 136 + ks * 32 + half * 16];
        v8bf a_lo = *(const v8bf*)arow;
        v8bf a_hi = *(const v8bf*)(arow + 8);
        v16bf afrag = __builtin_shufflevector(a_lo, a_hi,
            0, 1, 2, 3, 4, 5, 6, 7, 8, 9, 10, 11, 12, 13, 14, 15);
        // B fragment (32x16): lane col = n0+l15, contiguous 16 K from transposed Wc
        int kb = ks * 32 + half * 16;
        v16bf b0 = *(const v16bf*)(WcT + (long)(n0 + l15) * H_ + kb);
        v16bf b1 = *(const v16bf*)(WcT + (long)(n0 + 16 + l15) * H_ + kb);
        acc0 = __builtin_amdgcn_wmma_f32_16x16x32_bf16(false, afrag, false, b0,
                                                       (short)0, acc0, false, false);
        acc1 = __builtin_amdgcn_wmma_f32_16x16x32_bf16(false, afrag, false, b1,
                                                       (short)0, acc1, false, false);
    }
    // C layout: VGPR i -> row (i + half*8), col n0+l15 (tile0) / +16 (tile1)
    int col0 = n0 + l15;
    #pragma unroll
    for (int i = 0; i < 8; ++i) {
        long node = nodeRow[i + half * 8];
        z[node * H_ + col0]      = acc0[i];
        z[node * H_ + col0 + 16] = acc1[i];
    }
}

// ---------------- 3) el/er per (node, head) ------------------------------------
__global__ void attn_dots(const float* __restrict__ z,
                          const float* __restrict__ attn_l,
                          const float* __restrict__ attn_r,
                          float* __restrict__ el, float* __restrict__ er) {
    int idx = blockIdx.x * blockDim.x + threadIdx.x;
    if (idx >= N_ * NH_) return;
    int n = idx >> 2, h = idx & 3;
    const float* zr = z + (long)n * H_ + h * DH_;
    float sl = 0.f, sr = 0.f;
    #pragma unroll 8
    for (int d = 0; d < DH_; ++d) {
        float v = zr[d];
        sl += v * attn_l[h * DH_ + d];
        sr += v * attn_r[h * DH_ + d];
    }
    el[idx] = sl; er[idx] = sr;
}

// ---------------- 4a) per-dst max (encoded uint atomicMax) ---------------------
__global__ void edge_max(const int* __restrict__ src, const int* __restrict__ dst,
                         const float* __restrict__ el, const float* __restrict__ er,
                         unsigned* __restrict__ menc) {
    int e = blockIdx.x * blockDim.x + threadIdx.x;
    if (e >= E_) return;
    int s = src[e], d = dst[e];
    float4 l = *(const float4*)(el + (size_t)s * 4);
    float4 r = *(const float4*)(er + (size_t)d * 4);
    float v[4] = { l.x + r.x, l.y + r.y, l.z + r.z, l.w + r.w };
    #pragma unroll
    for (int h = 0; h < 4; ++h) {
        float ev = v[h] > 0.f ? v[h] : 0.2f * v[h];
        atomicMax(menc + (size_t)d * 4 + h, fenc(ev));
    }
}

// ---------------- 4b) per-dst softmax denominator ------------------------------
__global__ void edge_den(const int* __restrict__ src, const int* __restrict__ dst,
                         const float* __restrict__ el, const float* __restrict__ er,
                         const unsigned* __restrict__ menc, float* __restrict__ den) {
    int e = blockIdx.x * blockDim.x + threadIdx.x;
    if (e >= E_) return;
    int s = src[e], d = dst[e];
    float4 l = *(const float4*)(el + (size_t)s * 4);
    float4 r = *(const float4*)(er + (size_t)d * 4);
    float v[4] = { l.x + r.x, l.y + r.y, l.z + r.z, l.w + r.w };
    #pragma unroll
    for (int h = 0; h < 4; ++h) {
        float ev = v[h] > 0.f ? v[h] : 0.2f * v[h];
        float m  = fdec(menc[(size_t)d * 4 + h]);
        atomicAdd(den + (size_t)d * 4 + h, __expf(ev - m));
    }
}

// ---------------- 4c) wave-per-edge weighted scatter-add -----------------------
__global__ void edge_aggregate(const int* __restrict__ src, const int* __restrict__ dst,
                               const float* __restrict__ el, const float* __restrict__ er,
                               const unsigned* __restrict__ menc,
                               const float* __restrict__ den,
                               const float* __restrict__ z, float* __restrict__ agg) {
    long gt = (long)blockIdx.x * blockDim.x + threadIdx.x;
    int e = (int)(gt >> 5);
    if (e >= E_) return;
    int lane = (int)(gt & 31);
    int s = src[e], d = dst[e];
    int h = lane >> 3;                       // 8 lanes per head (4 floats each)
    float ev = el[(size_t)s * 4 + h] + er[(size_t)d * 4 + h];
    ev = ev > 0.f ? ev : 0.2f * ev;
    float m  = fdec(menc[(size_t)d * 4 + h]);
    float a  = __expf(ev - m) / (den[(size_t)d * 4 + h] + 1e-9f);
    int elem = lane * 4;
    float4 zv = *(const float4*)(z + (long)s * H_ + elem);
    float* ap = agg + (long)d * H_ + elem;
    atomicAdd(ap + 0, a * zv.x);
    atomicAdd(ap + 1, a * zv.y);
    atomicAdd(ap + 2, a * zv.z);
    atomicAdd(ap + 3, a * zv.w);
}

// ---------------- 5) ELU + linear scoring, wave-per-output ---------------------
__global__ void score_out(const float* __restrict__ agg,
                          const float* __restrict__ w_sent, const float* __restrict__ b_sent,
                          const float* __restrict__ w_feat, const float* __restrict__ b_feat,
                          float* __restrict__ out) {
    long gt = (long)blockIdx.x * blockDim.x + threadIdx.x;
    int w = (int)(gt >> 5);
    if (w >= B_ * (SP_ + FP_)) return;
    int lane = (int)(gt & 31);
    long node; const float* wv; float bias;
    if (w < B_ * SP_) {                       // sentence scores first
        int g = w / SP_, p = w % SP_;
        node = (long)g * NPG_ + FP_ + p;
        wv = w_sent; bias = b_sent[0];
    } else {                                  // then feature scores
        int w2 = w - B_ * SP_;
        int g = w2 / FP_, p = w2 % FP_;
        node = (long)g * NPG_ + p;
        wv = w_feat; bias = b_feat[0];
    }
    int elem = lane * 4;
    float4 a  = *(const float4*)(agg + node * H_ + elem);
    float4 ww = *(const float4*)(wv + elem);
    float hx = a.x > 0.f ? a.x : __expf(a.x) - 1.f;
    float hy = a.y > 0.f ? a.y : __expf(a.y) - 1.f;
    float hz = a.z > 0.f ? a.z : __expf(a.z) - 1.f;
    float hw = a.w > 0.f ? a.w : __expf(a.w) - 1.f;
    float sum = hx * ww.x + hy * ww.y + hz * ww.z + hw * ww.w;
    #pragma unroll
    for (int msk = 16; msk; msk >>= 1) sum += __shfl_xor(sum, msk, 32);
    if (lane == 0) out[w] = sum + bias;
}

// ---------------- host-side orchestration --------------------------------------
extern "C" void kernel_launch(void* const* d_in, const int* in_sizes, int n_in,
                              void* d_out, int out_size, void* d_ws, size_t ws_size,
                              hipStream_t stream) {
    const int*   fid      = (const int*)d_in[0];
    const int*   sid      = (const int*)d_in[1];
    const int*   uid      = (const int*)d_in[2];
    const int*   iid      = (const int*)d_in[3];
    const int*   esrc     = (const int*)d_in[4];
    const int*   edst     = (const int*)d_in[5];
    const float* user_tab = (const float*)d_in[6];
    const float* item_tab = (const float*)d_in[7];
    const float* feat_tab = (const float*)d_in[8];
    const float* sent_tab = (const float*)d_in[9];
    const float* W_feat   = (const float*)d_in[10];
    const float* W_sent   = (const float*)d_in[11];
    const float* W_user   = (const float*)d_in[12];
    const float* W_item   = (const float*)d_in[13];
    const float* W_gat    = (const float*)d_in[14];
    const float* attn_l   = (const float*)d_in[15];
    const float* attn_r   = (const float*)d_in[16];
    const float* w_sent   = (const float*)d_in[17];
    const float* b_sent   = (const float*)d_in[18];
    const float* w_feat   = (const float*)d_in[19];
    const float* b_feat   = (const float*)d_in[20];
    float* out = (float*)d_out;

    // workspace carve-out (256B aligned)
    uint8_t* ws = (uint8_t*)d_ws;
    size_t off = 0;
    auto carve = [&](size_t bytes) -> void* {
        void* p = ws + off;
        off = (off + bytes + 255) & ~(size_t)255;
        return p;
    };
    unsigned short* WcT  = (unsigned short*)carve((size_t)4 * H_ * H_ * 2);
    float*          z    = (float*)carve((size_t)N_ * H_ * 4);
    float*          el   = (float*)carve((size_t)N_ * NH_ * 4);
    float*          er   = (float*)carve((size_t)N_ * NH_ * 4);
    unsigned*       menc = (unsigned*)carve((size_t)N_ * NH_ * 4);
    float*          den  = (float*)carve((size_t)N_ * NH_ * 4);
    float*          agg  = (float*)carve((size_t)N_ * H_ * 4);

    hipMemsetAsync(menc, 0, (size_t)N_ * NH_ * 4, stream);
    hipMemsetAsync(den,  0, (size_t)N_ * NH_ * 4, stream);
    hipMemsetAsync(agg,  0, (size_t)N_ * H_ * 4, stream);

    // fold per-type projection into GAT projection, bf16 transposed
    combine_weights<<<(H_ * H_) / 256, 256, 0, stream>>>(W_feat, W_gat, WcT + 0 * H_ * H_);
    combine_weights<<<(H_ * H_) / 256, 256, 0, stream>>>(W_sent, W_gat, WcT + 1 * H_ * H_);
    combine_weights<<<(H_ * H_) / 256, 256, 0, stream>>>(W_user, W_gat, WcT + 2 * H_ * H_);
    combine_weights<<<(H_ * H_) / 256, 256, 0, stream>>>(W_item, W_gat, WcT + 3 * H_ * H_);

    // z = gather @ Wc per node type (all counts divisible by 16)
    node_project<<<(B_ * FP_) / 16, 128, 0, stream>>>(feat_tab, fid, WcT + 0 * H_ * H_, z, FP_, 0);
    node_project<<<(B_ * SP_) / 16, 128, 0, stream>>>(sent_tab, sid, WcT + 1 * H_ * H_, z, SP_, FP_);
    node_project<<<B_ / 16,          128, 0, stream>>>(user_tab, uid, WcT + 2 * H_ * H_, z, 1, FP_ + SP_);
    node_project<<<B_ / 16,          128, 0, stream>>>(item_tab, iid, WcT + 3 * H_ * H_, z, 1, FP_ + SP_ + 1);

    attn_dots<<<(N_ * NH_ + 255) / 256, 256, 0, stream>>>(z, attn_l, attn_r, el, er);

    edge_max<<<(E_ + 255) / 256, 256, 0, stream>>>(esrc, edst, el, er, menc);
    edge_den<<<(E_ + 255) / 256, 256, 0, stream>>>(esrc, edst, el, er, menc, den);

    long aggThreads = (long)E_ * 32;
    edge_aggregate<<<(int)((aggThreads + 255) / 256), 256, 0, stream>>>(
        esrc, edst, el, er, menc, den, z, agg);

    long scoreThreads = (long)B_ * (SP_ + FP_) * 32;
    score_out<<<(int)((scoreThreads + 255) / 256), 256, 0, stream>>>(
        agg, w_sent, b_sent, w_feat, b_feat, out);
}
